// RSNN2_47845935677487
// MI455X (gfx1250) — matrix operand order repaired and checked by reading, not compiled, not run on hardware
//
#include <hip/hip_runtime.h>
#include <hip/hip_bf16.h>

typedef __attribute__((ext_vector_type(16))) __bf16 v16bf;
typedef __attribute__((ext_vector_type(8)))  float  v8f;
typedef __attribute__((ext_vector_type(8)))  float  f8v;

#define B_    32
#define T_    500
#define NIN_  512
#define H_    2048
#define THR   (-55.0f)
#define RESTS 70.0f
#define BETA_ 0.95f
#define REFRAC 5

#define FRAG_U16 512  // one 16x32 (A) or 32x16 (B) bf16 fragment = 1KB = 512 u16

// ---- workspace layout (bytes) ----
#define CUR_BYTES  ((size_t)T_*B_*H_*4)          // 131.072 MB  cur[t][b][h] f32
#define WINP_OFF   CUR_BYTES
#define WINP_BYTES ((size_t)NIN_*H_*2)           // 2 MB packed bf16 W_in
#define WRECP_OFF  (WINP_OFF + WINP_BYTES)
#define WRECP_BYTES ((size_t)H_*H_*2)            // 8 MB packed bf16 W_rec
#define SPK_OFF    (WRECP_OFF + WRECP_BYTES)
#define SPK_U16    ((size_t)2*64*FRAG_U16)       // one spike buffer: 2 mt * 64 kt frags = 128 KB
#define BAR_OFF    (SPK_OFF + 2*SPK_U16*2 + 256)

// ---- scan kernel LDS chunking: 4 chunks of 16 k-tiles, 32 KB per chunk ----
#define CHUNK_KT  16
#define NCHUNK    (64 / CHUNK_KT)
#define LDS_BYTES (2 * CHUNK_KT * 1024)          // 32 KB (2 mt regions x 16 KB)

#if __has_builtin(__builtin_amdgcn_tensor_load_to_lds) && \
    __has_builtin(__builtin_amdgcn_s_wait_tensorcnt)
#define USE_TDM 1
#else
#define USE_TDM 0
#endif

#if USE_TDM
typedef __attribute__((ext_vector_type(4))) unsigned v4u;
typedef __attribute__((ext_vector_type(8))) int      v8i;
typedef __attribute__((ext_vector_type(4))) int      v4i;

// Issue a 1-D TDM load of `words` DWORDs from global -> LDS.
// D# group0: count=1, lds_addr, global_addr[56:0], type=2 (image).
// D# group1: data_size=4B (code 2), tensor_dim0=tile_dim0=words,
//            tensor_dim1=1, tensor_dim0_stride=words, no multicast/pad/iterate.
// This toolchain's builtin takes 6 args (groups 0..3 + extra group + cpol).
__device__ __forceinline__ void tdm_load_1d(unsigned lds_addr,
                                            unsigned long long gaddr,
                                            unsigned words) {
  v4u g0 = {1u,                                  // count=1 (valid user descriptor)
            lds_addr,                            // lds_addr [63:32]
            (unsigned)(gaddr & 0xFFFFFFFFull),   // global_addr[31:0]
            (unsigned)((gaddr >> 32) & 0x1FFFFFFull) | (2u << 30)}; // ga[56:32] | type=2
  v8i g1 = {(int)(2u << 16),                         // data_size=4B
            (int)((words & 0xFFFFu) << 16),          // tensor_dim0[15:0] at bits 63:48
            (int)(((words >> 16) & 0xFFFFu) | (1u << 16)), // dim0[31:16] | tensor_dim1=1
            (int)((words & 0xFFFFu) << 16),          // tile_dim0 at bits 127:112
            0,                                       // tile_dim1=0, tile_dim2=0 (unused)
            (int)words,                              // tensor_dim0_stride[31:0]
            0, 0};
  v4i z4 = {0, 0, 0, 0};
  v8i z8 = {0, 0, 0, 0, 0, 0, 0, 0};
  __builtin_amdgcn_tensor_load_to_lds(g0, g1, z4, z4, z8, 0);
}
#endif

// ---------------------------------------------------------------------------
// Kernel 1: pack weights into WMMA B-fragment layout (bf16), zero spk buf 0 +
// barrier. B-fragment (32x16 KxN bf16): lane<16 holds col n=lane, K {0..7,16..23};
// lane>=16 holds col n=lane-16, K {8..15,24..31} (2 bf16 per VGPR).
// ---------------------------------------------------------------------------
__global__ void rsnn_pack(const float* __restrict__ Win,
                          const float* __restrict__ Wrec,
                          __bf16* __restrict__ WinP,
                          __bf16* __restrict__ WrecP,
                          __bf16* __restrict__ spk0,
                          unsigned* __restrict__ bar) {
  size_t i = (size_t)blockIdx.x * blockDim.x + threadIdx.x;
  if (i < (size_t)H_ * H_) {             // pack W_rec: B[k][n] = W_rec[n][k]
    unsigned k = (unsigned)(i % H_), n = (unsigned)(i / H_);
    unsigned kt = k >> 5, kr = k & 31;
    unsigned nt = n >> 4, nr = n & 15;
    unsigned lane = nr + (((kr >> 3) & 1) << 4);
    unsigned e = (kr & 7) | ((kr >> 4) << 3);
    WrecP[((size_t)(kt * (H_ / 16) + nt)) * FRAG_U16 + lane * 16 + e] =
        (__bf16)Wrec[(size_t)n * H_ + k];
  }
  if (i < (size_t)NIN_ * H_) {           // pack W_in: B[k][n] = W_in[n][k]
    unsigned k = (unsigned)(i % NIN_), n = (unsigned)(i / NIN_);
    unsigned kt = k >> 5, kr = k & 31;
    unsigned nt = n >> 4, nr = n & 15;
    unsigned lane = nr + (((kr >> 3) & 1) << 4);
    unsigned e = (kr & 7) | ((kr >> 4) << 3);
    WinP[((size_t)(kt * (H_ / 16) + nt)) * FRAG_U16 + lane * 16 + e] =
        (__bf16)Win[(size_t)n * NIN_ + k];
  }
  if (i < SPK_U16) spk0[i] = (__bf16)0.0f;   // spk(t=0) == 0
  if (i < 2) bar[i * 16] = 0u;               // barrier count + generation
}

// ---------------------------------------------------------------------------
// Kernel 2: cur[t][b][h] = x[b][t][:] . W_in[h][:] + b_in[h]  via bf16 WMMA.
// One wave per 16x64 output strip. A-fragment (16x32 bf16) per ISA layout:
// lane<16: K {0..7,16..23}; lane>=16: K {8..15,24..31}; row M = lane%16.
// B fragments batched into registers before the 4 WMMAs -> partial loadcnt
// waits instead of s_wait_loadcnt 0 per WMMA.
// ---------------------------------------------------------------------------
__global__ void rsnn_input_gemm(const float* __restrict__ x,
                                const float* __restrict__ b_in,
                                const __bf16* __restrict__ WinP,
                                float* __restrict__ cur) {
  const unsigned lane = threadIdx.x & 31;
  const unsigned wave = threadIdx.x >> 5;
  const unsigned job = blockIdx.x * 8u + wave;  // 4000 blocks * 8 waves = 32000 jobs
  const unsigned mt = job >> 5;                 // 0..999  (16-row tile of M=T*B)
  const unsigned ng = job & 31;                 // 0..31   (group of 4 n-tiles)
  const unsigned t = mt >> 1;
  const unsigned bhalf = (mt & 1u) << 4;
  const unsigned lrow = lane & 15;
  const unsigned khalf = lane >> 4;
  const unsigned b = bhalf + lrow;

  v8f acc[4] = {{}, {}, {}, {}};
  const float* arow = x + ((size_t)b * T_ + t) * NIN_;

  for (unsigned kt = 0; kt < NIN_ / 32; ++kt) {
    const float* ap = arow + kt * 32 + khalf * 8;
    f8v alo = *(const f8v*)(ap);        // K k0+{0..7} (+8 if lane>=16)
    f8v ahi = *(const f8v*)(ap + 16);   // K k0+16+{0..7}
    const __bf16* bp = WinP + ((size_t)kt * (H_ / 16) + ng * 4u) * FRAG_U16 + lane * 16;
    v16bf bf4[4];
#pragma unroll
    for (int j = 0; j < 4; ++j) bf4[j] = *(const v16bf*)(bp + (size_t)j * FRAG_U16);
    v16bf a;
#pragma unroll
    for (int j = 0; j < 8; ++j) { a[j] = (__bf16)alo[j]; a[8 + j] = (__bf16)ahi[j]; }
#pragma unroll
    for (int j = 0; j < 4; ++j)
      acc[j] = __builtin_amdgcn_wmma_f32_16x16x32_bf16(false, a, false, bf4[j],
                                                       (short)0, acc[j], false, false);
  }
  // epilogue: C layout -> lane<16: M=v ; lane>=16: M=v+8 ; N = lane%16
  const unsigned rbase = khalf << 3;
#pragma unroll
  for (int j = 0; j < 4; ++j) {
    unsigned h = (ng * 4u + j) * 16u + lrow;
    float bias = b_in[h];
#pragma unroll
    for (int v = 0; v < 8; ++v) {
      unsigned bb = bhalf + rbase + v;
      cur[((size_t)t * B_ + bb) * H_ + h] = acc[j][v] + bias;
    }
  }
}

// ---------------------------------------------------------------------------
// Grid-wide barrier (32 co-resident workgroups, agent-scope fences).
// ---------------------------------------------------------------------------
__device__ __forceinline__ void grid_sync(unsigned* bar, unsigned nwg) {
  unsigned* cnt = bar;
  unsigned* gen = bar + 16;
  __syncthreads();
  if (threadIdx.x == 0) {
    __builtin_amdgcn_fence(__ATOMIC_RELEASE, "agent");
    unsigned g = __hip_atomic_load(gen, __ATOMIC_RELAXED, __HIP_MEMORY_SCOPE_AGENT);
    unsigned p = __hip_atomic_fetch_add(cnt, 1u, __ATOMIC_ACQ_REL, __HIP_MEMORY_SCOPE_AGENT);
    if (p == nwg - 1u) {
      __hip_atomic_store(cnt, 0u, __ATOMIC_RELAXED, __HIP_MEMORY_SCOPE_AGENT);
      __hip_atomic_fetch_add(gen, 1u, __ATOMIC_ACQ_REL, __HIP_MEMORY_SCOPE_AGENT);
    } else {
      while (__hip_atomic_load(gen, __ATOMIC_ACQUIRE, __HIP_MEMORY_SCOPE_AGENT) == g)
        __builtin_amdgcn_s_sleep(1);
    }
    __builtin_amdgcn_fence(__ATOMIC_ACQUIRE, "agent");
  }
  __syncthreads();
}

// ---------------------------------------------------------------------------
// Kernel 3: persistent sequential scan. 32 WGs x 8 waves; wave owns one 16x16
// tile (mt = batch-half, ntg = blockIdx*4 + wave%4) of the [32 x 2048] state.
// mem/refrac live in registers for all 500 steps. Per step: TDM stages the
// 128 KB packed spike buffer into LDS in 32 KB chunks (s_wait_tensorcnt),
// 64 bf16 WMMAs per wave (A from LDS via ds_load, B = W_rec from L2), fused
// LIF update on C-fragment registers, spike written to d_out (nontemporal)
// and to the next step's packed A-fragment buffer, then one grid barrier.
// ---------------------------------------------------------------------------
__global__ void rsnn_scan(const float* __restrict__ cur,
                          const float* __restrict__ mem0,
                          const float* __restrict__ b_rec,
                          const __bf16* __restrict__ WrecP,
                          __bf16* __restrict__ spkP,
                          float* __restrict__ out,
                          unsigned* __restrict__ bar) {
#if USE_TDM
  __shared__ __align__(128) unsigned char ldsA[LDS_BYTES];
  const unsigned ldsBase = (unsigned)(uintptr_t)(void*)ldsA;
#endif
  const unsigned lane = threadIdx.x & 31;
  const unsigned wave = threadIdx.x >> 5;
  const unsigned mt = wave >> 2;                      // 0..1 : batch half
  const unsigned ntg = blockIdx.x * 4u + (wave & 3);  // 0..127 : n-tile
  const unsigned lrow = lane & 15;
  const unsigned h = ntg * 16u + lrow;
  const unsigned rbase = (lane >> 4) << 3;

  float mem[8];
  int refr[8];
#pragma unroll
  for (int v = 0; v < 8; ++v) {
    unsigned b = mt * 16u + rbase + v;
    mem[v] = mem0[(size_t)b * H_ + h];
    refr[v] = 0;
  }
  const float brec = b_rec[h];

  // packed A-layout coordinates of this lane's spike elements (k = h fixed)
  const unsigned kr = h & 31, kt2 = h >> 5;
  const unsigned e2 = (kr & 7) | ((kr >> 4) << 3);
  const unsigned lanehalf = (kr >> 3) & 1;

  const unsigned nwg = H_ / 64;  // 32 workgroups

  for (unsigned t = 0; t < T_; ++t) {
    const __bf16* spkSrc = spkP + (size_t)(t & 1) * SPK_U16;
    __bf16* spkDst = spkP + (size_t)((t + 1) & 1) * SPK_U16;

    v8f acc[4] = {{}, {}, {}, {}};
    const __bf16* abase = spkSrc + (size_t)mt * 64 * FRAG_U16 + lane * 16;
    const __bf16* bbase = WrecP + (size_t)ntg * FRAG_U16 + lane * 16;
    (void)abase;

    for (unsigned c = 0; c < NCHUNK; ++c) {
#if USE_TDM
      __syncthreads();  // previous chunk's LDS reads complete before overwrite
      if (wave == 0) {  // stage 2 regions (mt=0/1, 16 k-tiles each) via TDM
#pragma unroll
        for (unsigned r = 0; r < 2; ++r)
          tdm_load_1d(ldsBase + r * (CHUNK_KT * 1024u),
                      (unsigned long long)(uintptr_t)spkSrc +
                          (unsigned long long)(r * 64u + c * CHUNK_KT) * 1024ull,
                      CHUNK_KT * 1024u / 4u);
        __builtin_amdgcn_s_wait_tensorcnt(0);
      }
      __syncthreads();  // staged spikes visible to all waves
#endif
#pragma unroll
      for (unsigned g = 0; g < CHUNK_KT; g += 4) {
        v16bf af[4], bf4[4];
#pragma unroll
        for (int j = 0; j < 4; ++j) {
          unsigned kt = c * CHUNK_KT + g + j;
#if USE_TDM
          af[j] = *(const v16bf*)(ldsA + (mt * CHUNK_KT + g + j) * 1024u + lane * 32u);
#else
          af[j] = *(const v16bf*)(abase + (size_t)kt * FRAG_U16);
#endif
          bf4[j] = *(const v16bf*)(bbase + (size_t)kt * (H_ / 16) * FRAG_U16);
        }
#pragma unroll
        for (int j = 0; j < 4; ++j)
          acc[j] = __builtin_amdgcn_wmma_f32_16x16x32_bf16(
              false, af[j], false, bf4[j], (short)0, acc[j], false, false);
      }
    }
    v8f rec = acc[0] + acc[1] + acc[2] + acc[3];

#pragma unroll
    for (int v = 0; v < 8; ++v) {
      unsigned b = mt * 16u + rbase + v;
      float cv = cur[((size_t)t * B_ + b) * H_ + h];
      if (t + 1 < T_)  // lowers to global_prefetch_b8 for next step's cur row
        __builtin_prefetch(&cur[((size_t)(t + 1) * B_ + b) * H_ + h], 0, 1);
      int rf = refr[v] - 1;
      rf = rf < 0 ? 0 : rf;
      float reset = (mem[v] - THR > 0.0f) ? 1.0f : 0.0f;                 // from old mem
      float base = BETA_ * (mem[v] + RESTS) + cv + rec[v] + brec - RESTS;
      float m2 = base * (1.0f - reset);                                   // zero-reset
      float spk = ((m2 - THR > 0.0f) && (rf == 0)) ? 1.0f : 0.0f;
      refr[v] = (spk > 0.0f) ? REFRAC : rf;
      mem[v] = m2;
      __builtin_nontemporal_store(spk, &out[((size_t)t * B_ + b) * H_ + h]);
      unsigned lane2 = (b & 15) + (lanehalf << 4);
      spkDst[((size_t)(mt * 64 + kt2)) * FRAG_U16 + lane2 * 16 + e2] = (__bf16)spk;
    }
    grid_sync(bar, nwg);
  }
}

// ---------------------------------------------------------------------------
extern "C" void kernel_launch(void* const* d_in, const int* in_sizes, int n_in,
                              void* d_out, int out_size, void* d_ws, size_t ws_size,
                              hipStream_t stream) {
  const float* x     = (const float*)d_in[0];
  const float* mem0  = (const float*)d_in[1];
  const float* W_in  = (const float*)d_in[2];
  const float* b_in  = (const float*)d_in[3];
  const float* W_rec = (const float*)d_in[4];
  const float* b_rec = (const float*)d_in[5];
  float* out = (float*)d_out;

  char* ws = (char*)d_ws;
  float*  cur   = (float*)ws;
  __bf16* WinP  = (__bf16*)(ws + WINP_OFF);
  __bf16* WrecP = (__bf16*)(ws + WRECP_OFF);
  __bf16* spkP  = (__bf16*)(ws + SPK_OFF);
  unsigned* bar = (unsigned*)(ws + BAR_OFF);

  // 1) pack weights to bf16 WMMA fragments, zero spike buffer + barrier
  rsnn_pack<<<(H_ * H_ + 255) / 256, 256, 0, stream>>>(W_in, W_rec, WinP, WrecP,
                                                       spkP, bar);
  // 2) input projection GEMM (16000 x 512 x 2048), bf16 WMMA
  rsnn_input_gemm<<<4000, 256, 0, stream>>>(x, b_in, WinP, cur);
  // 3) persistent 500-step recurrent scan, 32 co-resident WGs
  rsnn_scan<<<H_ / 64, 256, 0, stream>>>(cur, mem0, b_rec, WrecP, spkP, out, bar);
}
